// MPNN_Shortcuts_32684701122616
// MI455X (gfx1250) — matrix operand order repaired and checked by reading, not compiled, not run on hardware
//
#include <hip/hip_runtime.h>
#include <hip/hip_bf16.h>

// MPNN-with-shortcuts forward for MI455X (gfx1250, wave32).
// All dense GEMMs run on v_wmma_f32_16x16x32_f16 (f16 in, f32 accumulate).
// All B operands are stored [N][K] (transposed) so that both A and B WMMA
// fragments are contiguous 16B chunks -> 4 aligned b128 loads per WMMA,
// no per-iteration branches, minimal address math.
// APSP min-plus doubling is an LDS-tiled VALU kernel (min-plus is not a ring
// matmul, WMMA inapplicable). Working set ~35MB << 192MB L2 -> compute bound.

typedef _Float16 f16;
typedef __attribute__((ext_vector_type(16))) _Float16 v16h;
typedef __attribute__((ext_vector_type(8)))  _Float16 h8;   // 16B, align 16
typedef __attribute__((ext_vector_type(8)))  float    v8f;

#define NN 256
#define BBATCH 16
#define APSP_INF 1e9f

// ---------------- f32 -> f16 convert ----------------
__global__ __launch_bounds__(256) void k_cvt(const float* __restrict__ s,
                                             f16* __restrict__ d, long n) {
  long i = (long)blockIdx.x * 256 + threadIdx.x;
  if (i < n) d[i] = (f16)s[i];
}

// ---------------- f32 [R][C] -> f16 [C][R] (transpose convert) -------------
__global__ __launch_bounds__(256) void k_cvtT(const float* __restrict__ s,
                                              f16* __restrict__ d, int R, int C) {
  long i = (long)blockIdx.x * 256 + threadIdx.x;
  if (i >= (long)R * C) return;
  int r = (int)(i / C), c = (int)(i % C);
  d[(long)c * R + r] = (f16)s[i];
}

// ---------------- APSP: init D0 from adjacency ----------------
__global__ __launch_bounds__(256) void k_apsp_init(const float* __restrict__ adj,
                                                   float* __restrict__ D) {
  int b = blockIdx.y, i = blockIdx.x, j = threadIdx.x;
  const float* A = adj + (long)b * NN * NN;
  float a = fmaxf(A[(long)i * NN + j], A[(long)j * NN + i]);  // symmetrize
  D[((long)b * NN + i) * NN + j] = (i == j) ? 0.0f : (a > 0.0f ? 1.0f : APSP_INF);
}

// ---------------- APSP: one min-plus doubling step, 16x16 LDS tiles --------
__global__ __launch_bounds__(256) void k_apsp_step(const float* __restrict__ Din,
                                                   float* __restrict__ Dout) {
  int b = blockIdx.y;
  const float* D = Din + (long)b * NN * NN;
  float* O = Dout + (long)b * NN * NN;
  const int tiles = NN / 16;
  int ti = blockIdx.x / tiles, tj = blockIdx.x % tiles;
  int r = threadIdx.x >> 4, c = threadIdx.x & 15;
  int i = ti * 16 + r, k = tj * 16 + c;
  __shared__ float sA[16][16];
  __shared__ float sB[16][17];
  float best = 3.0e38f;
  for (int j0 = 0; j0 < NN; j0 += 16) {
    sA[r][c] = D[(long)i * NN + j0 + c];
    sB[r][c] = D[(long)(j0 + r) * NN + k];
    __syncthreads();
#pragma unroll
    for (int j = 0; j < 16; ++j) best = fminf(best, sA[r][j] + sB[j][c]);
    __syncthreads();
  }
  O[(long)i * NN + k] = best;
}

// ---------------- P_inv / Pb from final distances ----------------
__global__ __launch_bounds__(256) void k_pinv(const float* __restrict__ D,
                                              float* __restrict__ Pinv,
                                              float* __restrict__ Pb, long n) {
  long i = (long)blockIdx.x * 256 + threadIdx.x;
  if (i >= n) return;
  float P = D[i];
  if (P >= 0.5f * APSP_INF) P = 1.0f;            // where(isinf(P), 1.0)
  float pb = (P <= 10.0f) ? 0.0f : 1.0f;         // THRESH
  float pi = (P == 0.0f) ? 0.0f : 1.0f / (P * P);
  Pinv[i] = pi * pb;
  Pb[i] = pb;
}

// ---------------- row-normalized adjacency (f16 out) ----------------
__global__ __launch_bounds__(256) void k_adjnorm(const float* __restrict__ adj,
                                                 f16* __restrict__ adjn) {
  long row = blockIdx.x;  // b*NN + i
  const float* a = adj + row * NN;
  __shared__ float red[256];
  int t = threadIdx.x;
  red[t] = a[t];
  __syncthreads();
  for (int s = 128; s > 0; s >>= 1) {
    if (t < s) red[t] += red[t + s];
    __syncthreads();
  }
  float inv = 1.0f / (red[0] + 1.0f);  // deg + 1 (eye)
  adjn[row * NN + t] = (f16)(a[t] * inv);
}

// ---------------- sigmoid * Pinv, Gumbel hard argmax, adj2 ----------------
__global__ __launch_bounds__(256) void k_shortcut(const float* __restrict__ S,
                                                  const float* __restrict__ Pinv,
                                                  const float* __restrict__ Pb,
                                                  const float* __restrict__ gum,
                                                  const float* __restrict__ adj,
                                                  float* __restrict__ adj2, long n) {
  long i = (long)blockIdx.x * 256 + threadIdx.x;
  if (i >= n) return;
  float sm = (1.0f / (1.0f + expf(-S[i]))) * Pinv[i];
  float g0 = -logf(-logf(gum[2 * i]));
  float g1 = -logf(-logf(gum[2 * i + 1]));
  // forward value of straight-through gumbel: hard one-hot of argmax(logit+g);
  // ties pick index 0 -> '>='
  float y0 = (sm + g0 >= (1.0f - sm) + g1) ? 1.0f : 0.0f;
  adj2[i] = adj[i] + y0 * Pb[i];
}

// ---------------- mean over nodes: L[B,256,64] -> rep slice ----------------
__global__ __launch_bounds__(256) void k_mean(const f16* __restrict__ L,
                                              float* __restrict__ repF,
                                              f16* __restrict__ repH, int colOff) {
  int b = blockIdx.x >> 6;
  int col = blockIdx.x & 63;
  int t = threadIdx.x;
  __shared__ float red[256];
  red[t] = (float)L[((long)b * NN + t) * 64 + col];
  __syncthreads();
  for (int s = 128; s > 0; s >>= 1) {
    if (t < s) red[t] += red[t + s];
    __syncthreads();
  }
  if (t == 0) {
    float m = red[0] * (1.0f / 256.0f);
    repF[(long)b * 128 + colOff + col] = m;
    repH[(long)b * 128 + colOff + col] = (f16)m;
  }
}

// ---------------- final head: predict[b] = hid[b,:512] . w2 + b2 ----------
__global__ __launch_bounds__(256) void k_head2(const f16* __restrict__ hid,
                                               const f16* __restrict__ w2,
                                               const float* __restrict__ b2,
                                               float* __restrict__ pred) {
  int b = blockIdx.x, t = threadIdx.x;
  __shared__ float red[256];
  red[t] = (float)hid[(long)b * 512 + t] * (float)w2[t] +
           (float)hid[(long)b * 512 + 256 + t] * (float)w2[256 + t];
  __syncthreads();
  for (int s = 128; s > 0; s >>= 1) {
    if (t < s) red[t] += red[t + s];
    __syncthreads();
  }
  if (t == 0) pred[b] = red[0] + b2[0];
}

// ------------- generic WMMA GEMM: D = act(A[MxK] @ B[KxN] + bias) ----------
// B is stored TRANSPOSED: Bt[n*ldb + k]. One wave per 16x16 tile, 8 waves per
// block. Per the ISA f16 fragment layouts, with half = lane>>4, l15 = lane&15:
//   A frag  = A[row0+l15][k0 + half*8 + {0..7}] ++ A[..][k0+16+half*8+{0..7}]
//   B frag  = Bt[col0+l15][k0 + half*8 + {0..7}] ++ Bt[..][k0+16+half*8+{0..7}]
// i.e. 4 aligned 16-byte loads per WMMA (lda/ldb multiples of 16, k0 mult 32).
__global__ __launch_bounds__(256)
void wgemm16(const f16* __restrict__ A, int lda, long sA,
             const f16* __restrict__ Bt, int ldb, long sB,
             const float* __restrict__ bias,
             f16* __restrict__ outH, int ldoH, long sOH, int outHT,
             float* __restrict__ outF, int ldoF, long sOF,
             int M, int N, int K, int act) {
  int b = blockIdx.x;
  A += (long)b * sA;
  Bt += (long)b * sB;
  if (outH) outH += (long)b * sOH;
  if (outF) outF += (long)b * sOF;

  int wave = threadIdx.x >> 5;
  int lane = threadIdx.x & 31;
  int tilesN = N >> 4, tilesM = M >> 4;
  int tile = blockIdx.y * 8 + wave;
  if (tile >= tilesM * tilesN) return;  // wave-uniform: EXEC stays all-ones
  int row0 = (tile / tilesN) << 4;
  int col0 = (tile % tilesN) << 4;

  int half = lane >> 4;   // lane group
  int l15 = lane & 15;    // A row within tile == output column within tile

  const f16* Ap = A + (long)(row0 + l15) * lda + (half << 3);
  const f16* Bp = Bt + (long)(col0 + l15) * ldb + (half << 3);

  union V16 { v16h v; h8 p[2]; };
  v8f c = {};
#pragma unroll 2
  for (int k0 = 0; k0 < K; k0 += 32) {
    V16 a, bf;
    a.p[0] = *(const h8*)(Ap + k0);        // K = k0 + half*8 + 0..7
    a.p[1] = *(const h8*)(Ap + k0 + 16);   // K = k0 + 16 + half*8 + 0..7
    bf.p[0] = *(const h8*)(Bp + k0);       // K = k0 + half*8 + 0..7
    bf.p[1] = *(const h8*)(Bp + k0 + 16);  // K = k0 + 16 + half*8 + 0..7
    c = __builtin_amdgcn_wmma_f32_16x16x32_f16(false, a.v, false, bf.v,
                                               (short)0, c, false, false);
  }

  float bv = bias ? bias[col0 + l15] : 0.0f;
#pragma unroll
  for (int r = 0; r < 8; ++r) {
    float x = c[r] + bv;
    if (act == 1) x = tanhf(x);
    long m = row0 + r + (half << 3);
    long n = col0 + l15;
    if (outH) {
      if (outHT) outH[n * ldoH + m] = (f16)x;   // write output transposed
      else       outH[m * ldoH + n] = (f16)x;
    }
    if (outF) outF[m * ldoF + n] = x;
  }
}

// =====================================================================
struct EncP {  // f32 device pointers for one encoder
  const float *nfc1_w, *nfc1_b, *nfc2_w, *nfc2_b, *conv_w;
  const float *lfc1_w, *lfc1_b, *lfc2_w, *lfc2_b;
};

extern "C" void kernel_launch(void* const* d_in, const int* in_sizes, int n_in,
                              void* d_out, int out_size, void* d_ws, size_t ws_size,
                              hipStream_t stream) {
  if (n_in < 25) return;
  auto F = [&](int i) { return (const float*)d_in[i]; };

  // ---- input mapping: detect insertion-order vs JAX sorted-key flattening
  bool ins = (in_sizes[1] == BBATCH * NN * 32);  // node_feat at index 1?
  const float *adj, *node_feat, *gum, *fc1_w, *fc1_b, *fc2_w, *fc2_b;
  EncP E1, E2;
  auto getEnc = [&](int base) {
    EncP E;
    if (ins) {  // nfc1_w, nfc1_b, nfc2_w, nfc2_b, conv_w, lfc1_w, lfc1_b, lfc2_w, lfc2_b
      E.nfc1_w = F(base); E.nfc1_b = F(base + 1); E.nfc2_w = F(base + 2);
      E.nfc2_b = F(base + 3); E.conv_w = F(base + 4); E.lfc1_w = F(base + 5);
      E.lfc1_b = F(base + 6); E.lfc2_w = F(base + 7); E.lfc2_b = F(base + 8);
    } else {    // sorted: conv_w, lfc1_b, lfc1_w, lfc2_b, lfc2_w, nfc1_b, nfc1_w, nfc2_b, nfc2_w
      E.conv_w = F(base); E.lfc1_b = F(base + 1); E.lfc1_w = F(base + 2);
      E.lfc2_b = F(base + 3); E.lfc2_w = F(base + 4); E.nfc1_b = F(base + 5);
      E.nfc1_w = F(base + 6); E.nfc2_b = F(base + 7); E.nfc2_w = F(base + 8);
    }
    return E;
  };
  if (ins) {
    adj = F(0); node_feat = F(1); gum = F(2);
    E1 = getEnc(3); E2 = getEnc(12);
    fc1_w = F(21); fc1_b = F(22); fc2_w = F(23); fc2_b = F(24);
  } else {    // sorted top-level: adj, enc1, enc2, gumbel_u, head, node_feat
    adj = F(0); E1 = getEnc(1); E2 = getEnc(10); gum = F(19);
    fc1_b = F(20); fc1_w = F(21); fc2_b = F(22); fc2_w = F(23); node_feat = F(24);
  }

  // ---- workspace carving
  size_t off = 0;
  char* w = (char*)d_ws;
  auto carve = [&](size_t bytes) {
    void* p = w + off;
    off += (bytes + 255) & ~(size_t)255;
    return p;
  };
  const long NN2 = (long)BBATCH * NN * NN;  // 1,048,576
  float* Da   = (float*)carve(NN2 * 4);
  float* Db   = (float*)carve(NN2 * 4);     // reused as S (scores) after APSP
  float* Pinv = (float*)carve(NN2 * 4);
  float* Pb   = (float*)carve(NN2 * 4);
  float* adj2 = (float*)carve(NN2 * 4);
  f16* adjn1  = (f16*)carve(NN2 * 2);
  f16* adjn2  = (f16*)carve(NN2 * 2);
  f16* nf16   = (f16*)carve((long)BBATCH * NN * 32 * 2);
  f16* H0     = (f16*)carve(NN2 * 2);                       // [B,256,256]
  f16* allh   = (f16*)carve((long)BBATCH * NN * 320 * 2);   // [B,256,320]
  f16* XWT    = (f16*)carve((long)BBATCH * NN * 64 * 2);    // [B,64,256] (T)
  f16* L1     = (f16*)carve((long)BBATCH * NN * 64 * 2);
  f16* L2     = (f16*)carve((long)BBATCH * NN * 64 * 2);
  f16* repH   = (f16*)carve(BBATCH * 128 * 2);
  f16* hid    = (f16*)carve(BBATCH * 512 * 2);
  f16* w_n1 = (f16*)carve(32 * 256 * 2);    // [256][32]   (T)
  f16* w_n2 = (f16*)carve(256 * 64 * 2);    // [64][256]   (T)
  f16* w_cw = (f16*)carve(4 * 64 * 64 * 2); // 4x [64][64] (T)
  f16* w_l1 = (f16*)carve(320 * 256 * 2);   // [256][320]  (T)
  f16* w_l2 = (f16*)carve(256 * 64 * 2);    // [64][256]   (T)
  f16* w_h1 = (f16*)carve(128 * 512 * 2);   // [512][128]  (T)
  f16* w_h2 = (f16*)carve(512 * 2);
  (void)ws_size; (void)out_size;

  auto cvt = [&](const float* s, f16* d, long n) {
    k_cvt<<<dim3((unsigned)((n + 255) / 256)), dim3(256), 0, stream>>>(s, d, n);
  };
  auto cvtT = [&](const float* s, f16* d, int R, int C) {
    long n = (long)R * C;
    k_cvtT<<<dim3((unsigned)((n + 255) / 256)), dim3(256), 0, stream>>>(s, d, R, C);
  };
  auto gemm = [&](const f16* A, int lda, long sA, const f16* Bt, int ldb, long sB,
                  const float* bias, f16* oH, int ldoH, long sOH, int oHT,
                  float* oF, int ldoF, long sOF, int M, int N, int K, int act,
                  int batches) {
    int tiles = (M / 16) * (N / 16);
    dim3 g(batches, (tiles + 7) / 8);
    wgemm16<<<g, dim3(256), 0, stream>>>(A, lda, sA, Bt, ldb, sB, bias,
                                         oH, ldoH, sOH, oHT, oF, ldoF, sOF,
                                         M, N, K, act);
  };

  // ======== APSP: min-plus doubling (8 = ceil(log2(256)) steps) ========
  k_apsp_init<<<dim3(NN, BBATCH), dim3(256), 0, stream>>>(adj, Da);
  float *cur = Da, *nxt = Db;
  for (int it = 0; it < 8; ++it) {
    k_apsp_step<<<dim3((NN / 16) * (NN / 16), BBATCH), dim3(256), 0, stream>>>(cur, nxt);
    float* t = cur; cur = nxt; nxt = t;
  }
  // 8 swaps -> result in Da (cur); Db (nxt) is free -> reuse as score buffer
  float* S = nxt;
  k_pinv<<<dim3((unsigned)((NN2 + 255) / 256)), dim3(256), 0, stream>>>(cur, Pinv, Pb, NN2);

  // ======== common conversions ========
  cvt(node_feat, nf16, (long)BBATCH * NN * 32);
  k_adjnorm<<<dim3(BBATCH * NN), dim3(256), 0, stream>>>(adj, adjn1);

  // ======== encoder (12 WMMA GEMMs each) ========
  auto runEnc = [&](const EncP& E, const f16* adjn, f16* Lout) {
    cvtT(E.nfc1_w, w_n1, 32, 256);
    cvtT(E.nfc2_w, w_n2, 256, 64);
    for (int l = 0; l < 4; ++l) cvtT(E.conv_w + l * 4096, w_cw + l * 4096, 64, 64);
    cvtT(E.lfc1_w, w_l1, 320, 256);
    cvtT(E.lfc2_w, w_l2, 256, 64);
    // H0 = tanh(X @ nfc1_w + b)            [256x32]@[32x256]
    gemm(nf16, 32, (long)NN * 32, w_n1, 32, 0, E.nfc1_b,
         H0, 256, (long)NN * 256, 0, nullptr, 0, 0, NN, 256, 32, 1, BBATCH);
    // allh[:,0:64] = tanh(H0 @ nfc2_w + b) [256x256]@[256x64]
    gemm(H0, 256, (long)NN * 256, w_n2, 256, 0, E.nfc2_b,
         allh, 320, (long)NN * 320, 0, nullptr, 0, 0, NN, 64, 256, 1, BBATCH);
    for (int l = 0; l < 4; ++l) {
      // XW^T = (cur @ conv_w[l])^T         [256x64]@[64x64], epilogue-transposed
      gemm(allh + l * 64, 320, (long)NN * 320, w_cw + l * 4096, 64, 0, nullptr,
           XWT, 256, (long)NN * 64, 1, nullptr, 0, 0, NN, 64, 64, 0, BBATCH);
      // cur' = tanh(adj_n @ XW)            [256x256]@[256x64], B given as XW^T
      gemm(adjn, 256, (long)NN * 256, XWT, 256, (long)NN * 64, nullptr,
           allh + (l + 1) * 64, 320, (long)NN * 320, 0, nullptr, 0, 0,
           NN, 64, 256, 1, BBATCH);
    }
    // lat = tanh(allh @ lfc1 + b)          [256x320]@[320x256] (reuse H0)
    gemm(allh, 320, (long)NN * 320, w_l1, 320, 0, E.lfc1_b,
         H0, 256, (long)NN * 256, 0, nullptr, 0, 0, NN, 256, 320, 1, BBATCH);
    // L = tanh(lat @ lfc2 + b)             [256x256]@[256x64]
    gemm(H0, 256, (long)NN * 256, w_l2, 256, 0, E.lfc2_b,
         Lout, 64, (long)NN * 64, 0, nullptr, 0, 0, NN, 64, 256, 1, BBATCH);
  };

  runEnc(E1, adjn1, L1);

  // ======== S = L1 @ L1^T (L1 is already [N][K] for B), then shortcuts =====
  gemm(L1, 64, (long)NN * 64, L1, 64, (long)NN * 64, nullptr,
       nullptr, 0, 0, 0, S, 256, (long)NN * 256, NN, 256, 64, 0, BBATCH);
  k_shortcut<<<dim3((unsigned)((NN2 + 255) / 256)), dim3(256), 0, stream>>>(
      S, Pinv, Pb, gum, adj, adj2, NN2);
  k_adjnorm<<<dim3(BBATCH * NN), dim3(256), 0, stream>>>(adj2, adjn2);

  runEnc(E2, adjn2, L2);

  // ======== rep = [mean(L1), mean(L2)] -> d_out[16:] and f16 for head ====
  float* outF = (float*)d_out;          // [0:16) predict, [16:2064) rep
  k_mean<<<dim3(BBATCH * 64), dim3(256), 0, stream>>>(L1, outF + 16, repH, 0);
  k_mean<<<dim3(BBATCH * 64), dim3(256), 0, stream>>>(L2, outF + 16, repH, 64);

  // ======== head: hid = tanh(rep @ fc1 + b) (WMMA), predict = hid @ fc2 + b
  cvtT(fc1_w, w_h1, 128, 512);
  cvt(fc2_w, w_h2, 512);
  gemm(repH, 128, 0, w_h1, 128, 0, fc1_b,
       hid, 512, 0, 0, nullptr, 0, 0, 16, 512, 128, 1, 1);
  k_head2<<<dim3(BBATCH), dim3(256), 0, stream>>>(hid, w_h2, fc2_b, outF);
}
